// DeepseekV2MoE_8048768713516
// MI455X (gfx1250) — compile-verified, matrix operand order checked by reading, not run on hardware
//
#include <hip/hip_runtime.h>
#include <hip/hip_bf16.h>

// DeepseekV2 MoE for MI455X (gfx1250, wave32, WMMA).
// fp32 weights streamed once (BW floor ~27us @23.3TB/s), converted to f16 in
// registers, matmuls on v_wmma_f32_16x16x32_f16 with fp32 accumulators.
// K-loops are software-pipelined: next iteration's raw float4 loads are issued
// before the current iteration's converts/WMMAs, so VMEM latency overlaps the
// matrix pipe (XDL WMMA co-executes with VALU converts on CDNA5).

typedef __attribute__((ext_vector_type(16))) _Float16 v16h;
typedef __attribute__((ext_vector_type(8)))  float    v8f;

#define T_TOK   1024
#define DMODEL  2048
#define IEXP    1408
#define NEXP    16
#define SSHARED 2816   // shared intermediate (2*I)
#define TOPK    6
#define NGROUP  4
#define TOPKG   2
#define RSCALE  2.5f

// Fast sigmoid: one v_exp_f32 + one v_rcp_f32 (no IEEE divide sequence).
__device__ __forceinline__ float fast_sigmoid(float v) {
  return __builtin_amdgcn_rcpf(1.f + __expf(-v));
}

// Raw (fp32) fragment: 16 floats per lane, loaded as 4x global_load_b128 from a
// row pointer already offset by (lane>=16 ? 8 : 0). ISA layout (CDNA5 ISA):
//   lanes 0-15 : K = {k0..k0+7} in v0..v3, {k0+16..k0+23} in v4..v7
//   lanes 16-31: K = {k0+8..k0+15},        {k0+24..k0+31}
struct RawFrag { float4 f0, f1, f2, f3; };

__device__ __forceinline__ RawFrag load_raw(const float* __restrict__ p) {
  RawFrag r;
  r.f0 = *reinterpret_cast<const float4*>(p);
  r.f1 = *reinterpret_cast<const float4*>(p + 4);
  r.f2 = *reinterpret_cast<const float4*>(p + 16);
  r.f3 = *reinterpret_cast<const float4*>(p + 20);
  return r;
}

__device__ __forceinline__ v16h cvt_frag(const RawFrag& f) {
  v16h r;
  r[0]  = (_Float16)f.f0.x; r[1]  = (_Float16)f.f0.y;
  r[2]  = (_Float16)f.f0.z; r[3]  = (_Float16)f.f0.w;
  r[4]  = (_Float16)f.f1.x; r[5]  = (_Float16)f.f1.y;
  r[6]  = (_Float16)f.f1.z; r[7]  = (_Float16)f.f1.w;
  r[8]  = (_Float16)f.f2.x; r[9]  = (_Float16)f.f2.y;
  r[10] = (_Float16)f.f2.z; r[11] = (_Float16)f.f2.w;
  r[12] = (_Float16)f.f3.x; r[13] = (_Float16)f.f3.y;
  r[14] = (_Float16)f.f3.z; r[15] = (_Float16)f.f3.w;
  return r;
}

__global__ void zero_kernel(float* __restrict__ p, int n) {
  int i = blockIdx.x * blockDim.x + threadIdx.x;
  if (i < n) p[i] = 0.f;
}

// One block (256 threads) per token: 16 logits, sigmoid scoring + bias,
// grouped top-2-of-4-groups, masked top-6, renormalized sigmoid weights.
__global__ void moe_route_kernel(const float* __restrict__ x,
                                 const float* __restrict__ gate_w,
                                 const float* __restrict__ bias,
                                 float* __restrict__ combine) {
  __shared__ float part[NEXP][17];
  __shared__ float logits[NEXP];
  const int t = blockIdx.x;
  const int e = threadIdx.x >> 4;   // expert 0..15
  const int j = threadIdx.x & 15;   // 16 threads per expert
  const float* xr = x + (size_t)t * DMODEL;
  const float* gr = gate_w + (size_t)e * DMODEL;
  float s = 0.f;
  for (int k = j; k < DMODEL; k += 16) s += xr[k] * gr[k];
  part[e][j] = s;
  __syncthreads();
  if (threadIdx.x < NEXP) {
    float acc = 0.f;
    #pragma unroll
    for (int i = 0; i < 16; ++i) acc += part[threadIdx.x][i];
    logits[threadIdx.x] = acc;
  }
  __syncthreads();
  if (threadIdx.x == 0) {
    float scores[NEXP], sc[NEXP];
    #pragma unroll
    for (int i = 0; i < NEXP; ++i) {
      float sg = fast_sigmoid(logits[i]);
      scores[i] = sg;
      sc[i] = sg + bias[i];
    }
    const int eg = NEXP / NGROUP;                 // 4 experts per group
    float gs[NGROUP];
    #pragma unroll
    for (int g = 0; g < NGROUP; ++g) {            // sum of top-2 in group
      float m1 = -__builtin_inff(), m2 = -__builtin_inff();
      #pragma unroll
      for (int i = 0; i < 4; ++i) {
        float v = sc[g * eg + i];
        if (v > m1) { m2 = m1; m1 = v; } else if (v > m2) { m2 = v; }
      }
      gs[g] = m1 + m2;
    }
    bool gsel[NGROUP] = {false, false, false, false};
    for (int k = 0; k < TOPKG; ++k) {             // top-2 groups, low idx wins tie
      int best = 0; float bv = -__builtin_inff();
      for (int g = 0; g < NGROUP; ++g)
        if (!gsel[g] && gs[g] > bv) { bv = gs[g]; best = g; }
      gsel[best] = true;
    }
    bool used[NEXP];
    for (int i = 0; i < NEXP; ++i) used[i] = false;
    int idx[TOPK]; float wv[TOPK]; float wsum = 0.f;
    for (int k = 0; k < TOPK; ++k) {              // masked top-6, low idx wins tie
      int best = 0; float bv = -__builtin_inff();
      for (int i = 0; i < NEXP; ++i) {
        if (used[i] || !gsel[i / eg]) continue;
        if (sc[i] > bv) { bv = sc[i]; best = i; }
      }
      used[best] = true;
      idx[k] = best;
      wv[k] = scores[best];
      wsum += scores[best];
    }
    const float winv = __builtin_amdgcn_rcpf(wsum);
    float* crow = combine + (size_t)t * NEXP;
    for (int i = 0; i < NEXP; ++i) crow[i] = 0.f;
    for (int k = 0; k < TOPK; ++k) crow[idx[k]] = wv[k] * winv;
  }
}

// h[m, n] = rowscale(m) * silu(x[m,:] . wgu[n,:]) * (x[m,:] . wgu[Isz+n,:])
// Block = 8 waves; wave w owns output columns n0 = bx*128 + w*16 (two WMMA
// accumulator chains: gate + up). K loop over DMODEL in steps of 32,
// double-buffered. HAS_SCALE: scale row m by combine[m*NEXP]*RSCALE.
template <bool HAS_SCALE>
__global__ void __launch_bounds__(256)
gateup_silu_kernel(const float* __restrict__ x,
                   const float* __restrict__ wgu,      // [2*Isz, DMODEL]
                   float* __restrict__ h,              // [T, Isz]
                   const float* __restrict__ rowscale, // combine+e, stride NEXP
                   int Isz) {
  const int lane = threadIdx.x & 31;
  const int wave = threadIdx.x >> 5;
  const int hi   = lane >> 4;
  const int l15  = lane & 15;
  const int n0   = blockIdx.x * 128 + wave * 16;
  const int m0   = blockIdx.y * 16;
  const int koff = hi ? 8 : 0;
  const float* arow = x   + (size_t)(m0 + l15) * DMODEL + koff;
  const float* grow = wgu + (size_t)(n0 + l15) * DMODEL + koff;
  const float* urow = wgu + (size_t)(Isz + n0 + l15) * DMODEL + koff;
  v8f cg = {}; v8f cu = {};
  RawFrag ra = load_raw(arow);
  RawFrag rg = load_raw(grow);
  RawFrag ru = load_raw(urow);
  for (int k0 = 0; k0 < DMODEL; k0 += 32) {
    // clamp so the (redundant) last-iteration prefetch stays in bounds
    const int kn = (k0 + 32 < DMODEL) ? k0 + 32 : k0;
    __builtin_prefetch(grow + k0 + 512, 0, 0);   // stream weights ahead
    __builtin_prefetch(urow + k0 + 512, 0, 0);
    RawFrag na = load_raw(arow + kn);
    RawFrag ng = load_raw(grow + kn);
    RawFrag nu = load_raw(urow + kn);
    v16h a  = cvt_frag(ra);
    v16h bg = cvt_frag(rg);
    v16h bu = cvt_frag(ru);
    cg = __builtin_amdgcn_wmma_f32_16x16x32_f16(false, a, false, bg, (short)0, cg, false, false);
    cu = __builtin_amdgcn_wmma_f32_16x16x32_f16(false, a, false, bu, (short)0, cu, false, false);
    ra = na; rg = ng; ru = nu;
  }
  // Hoist the 8 row scales into one load clause (no per-element branch/stall).
  float sarr[8];
  #pragma unroll
  for (int i = 0; i < 8; ++i) {
    sarr[i] = HAS_SCALE ? rowscale[(m0 + i + 8 * hi) * NEXP] * RSCALE : 1.f;
  }
  // C/D layout: VGPR i, lane L -> M = i + 8*(L>=16), N = L&15
  const int col = n0 + l15;
  #pragma unroll
  for (int i = 0; i < 8; ++i) {
    const int row = m0 + i + 8 * hi;
    float g = cg[i], u = cu[i];
    float val = g * fast_sigmoid(g) * u;        // silu(g)*u
    h[(size_t)row * Isz + col] = val * sarr[i];
  }
}

// out[m, n] += h[m,:] . wd[n,:]   (wd is [DMODEL, K], out is [T, DMODEL])
__global__ void __launch_bounds__(256)
down_acc_kernel(const float* __restrict__ h,   // [T, K]
                const float* __restrict__ wd,  // [DMODEL, K]
                float* __restrict__ out,       // [T, DMODEL]
                int K) {
  __builtin_assume(K >= 32 && (K & 31) == 0);
  const int lane = threadIdx.x & 31;
  const int wave = threadIdx.x >> 5;
  const int hi   = lane >> 4;
  const int l15  = lane & 15;
  const int n0   = blockIdx.x * 128 + wave * 16;
  const int m0   = blockIdx.y * 16;
  const int koff = hi ? 8 : 0;
  const float* arow = h  + (size_t)(m0 + l15) * K + koff;
  const float* brow = wd + (size_t)(n0 + l15) * K + koff;
  v8f c = {};
  RawFrag ra = load_raw(arow);
  RawFrag rb = load_raw(brow);
  for (int k0 = 0; k0 < K; k0 += 32) {
    const int kn = (k0 + 32 < K) ? k0 + 32 : k0;
    __builtin_prefetch(brow + k0 + 512, 0, 0);
    RawFrag na = load_raw(arow + kn);
    RawFrag nb = load_raw(brow + kn);
    v16h a = cvt_frag(ra);
    v16h b = cvt_frag(rb);
    c = __builtin_amdgcn_wmma_f32_16x16x32_f16(false, a, false, b, (short)0, c, false, false);
    ra = na; rb = nb;
  }
  const int col = n0 + l15;
  #pragma unroll
  for (int i = 0; i < 8; ++i) {
    const int row = m0 + i + 8 * hi;
    out[(size_t)row * DMODEL + col] += c[i];
  }
}

extern "C" void kernel_launch(void* const* d_in, const int* in_sizes, int n_in,
                              void* d_out, int out_size, void* d_ws, size_t ws_size,
                              hipStream_t stream) {
  (void)in_sizes; (void)n_in; (void)out_size; (void)ws_size;
  const float* x          = (const float*)d_in[0];
  const float* gate_w     = (const float*)d_in[1];
  const float* bias       = (const float*)d_in[2];
  const float* w_gate_up  = (const float*)d_in[3];
  const float* w_down     = (const float*)d_in[4];
  const float* shared_wgu = (const float*)d_in[5];
  const float* shared_wd  = (const float*)d_in[6];
  float* out = (float*)d_out;

  // workspace: combine [T,E] (64KB) + hidden buffer [T, SSHARED] (11.5MB, reused)
  float* wsf     = (float*)d_ws;
  float* combine = wsf;
  float* hbuf    = wsf + (size_t)T_TOK * NEXP;

  zero_kernel<<<(T_TOK * DMODEL + 255) / 256, 256, 0, stream>>>(out, T_TOK * DMODEL);
  moe_route_kernel<<<T_TOK, 256, 0, stream>>>(x, gate_w, bias, combine);

  // shared expert: out += silu_mul(x @ Wgu^T) @ Wd^T (unscaled)
  gateup_silu_kernel<false><<<dim3(SSHARED / 128, T_TOK / 16), 256, 0, stream>>>(
      x, shared_wgu, hbuf, nullptr, SSHARED);
  down_acc_kernel<<<dim3(DMODEL / 128, T_TOK / 16), 256, 0, stream>>>(
      hbuf, shared_wd, out, SSHARED);

  // routed experts: h scaled by combine[t,e]*2.5 before down-proj, so
  // unrouted rows contribute exactly zero (matches reference's dense scan).
  for (int e = 0; e < NEXP; ++e) {
    const float* wgu_e = w_gate_up + (size_t)e * 2 * IEXP * DMODEL;
    const float* wd_e  = w_down    + (size_t)e * DMODEL * IEXP;
    gateup_silu_kernel<true><<<dim3(IEXP / 128, T_TOK / 16), 256, 0, stream>>>(
        x, wgu_e, hbuf, combine + e, IEXP);
    down_acc_kernel<<<dim3(DMODEL / 128, T_TOK / 16), 256, 0, stream>>>(
        hbuf, wd_e, out, IEXP);
  }
}
